// SelfAttentionBlock2D_58798102282696
// MI455X (gfx1250) — matrix operand order repaired and checked by reading, not compiled, not run on hardware
//
#include <hip/hip_runtime.h>
#include <cstdint>
#include <cstddef>

// ---------------------------------------------------------------------------
// Problem constants (from reference: B=8, Cin=512, H=W=64, Ck=Cv=256, Cout=512)
// ---------------------------------------------------------------------------
#define B_    8
#define CIN   512
#define CK    256
#define CV    256
#define COUT  512
#define NPIX  4096   // H*W
#define KTILE 32     // keys staged per attention iteration

typedef __bf16 bf16_t;
typedef bf16_t bf16x16 __attribute__((ext_vector_type(16)));
typedef float  f32x8   __attribute__((ext_vector_type(8)));
typedef unsigned int u32x4 __attribute__((ext_vector_type(4)));
typedef int          i32x4 __attribute__((ext_vector_type(4)));
typedef int          i32x8 __attribute__((ext_vector_type(8)));

union FragU  { uint4 q[2]; bf16x16 m; };          // 32B A/B fragment (16 bf16)
union PFragU { unsigned int u[8]; bf16x16 m; };   // packed P^T fragment

__device__ __forceinline__ unsigned short f2bf(float f) {
    unsigned int u = __float_as_uint(f);
    unsigned int r = u + 0x7fffu + ((u >> 16) & 1u);   // round-to-nearest-even
    return (unsigned short)(r >> 16);
}
__device__ __forceinline__ unsigned int packbf(float a, float b) {
    return (unsigned int)f2bf(a) | ((unsigned int)f2bf(b) << 16);
}
// lane <-> lane^16 exchange (wave32): ds_swizzle group-of-32, xor=0x10,and=0x1f
__device__ __forceinline__ float swapx16(float x) {
    return __int_as_float(__builtin_amdgcn_ds_swizzle(__float_as_int(x), 0x401f));
}
__device__ __forceinline__ f32x8 wmma_bf16(bf16x16 a, bf16x16 b, f32x8 c) {
    return __builtin_amdgcn_wmma_f32_16x16x32_bf16(false, a, false, b, (short)0, c,
                                                   false, false);
}

// ---------------------------------------------------------------------------
// Tensor Data Mover: 2D tile (rows x d0u 8-byte units) Global -> LDS.
// D# per CDNA5 ISA 8.3/8.4: group0 = {count|flags, lds_addr, gaddr lo, gaddr hi|type=2},
// group1 = {data_size, tensor_dim0/1, tile_dim0/1/2, dim0 stride}; groups 2/3 zero (2-D).
// Issued by one wave; completion via TENSORcnt.
// ---------------------------------------------------------------------------
__device__ __forceinline__ void tdm_load_2d(unsigned lds_off, const unsigned short* g,
                                            unsigned d0u, unsigned rows, unsigned strideu)
{
    unsigned long long ga = (unsigned long long)(uintptr_t)g;
    u32x4 g0 = { 1u,                                            // count=1 (valid user D#)
                 lds_off,                                       // lds_addr (bytes)
                 (unsigned)(ga & 0xffffffffull),                // global_addr[31:0]
                 (unsigned)((ga >> 32) & 0x1ffffffull) | (2u << 30) }; // [56:32] | type=2
    i32x8 g1 = { (int)(3u << 16),                               // data_size = 8B units
                 (int)((d0u & 0xffffu) << 16),                  // tensor_dim0[15:0]
                 (int)(((d0u >> 16) & 0xffffu) | ((rows & 0xffffu) << 16)), // dim0 hi | dim1 lo
                 (int)(((rows >> 16) & 0xffffu) | ((d0u & 0xffffu) << 16)), // dim1 hi | tile_dim0
                 (int)(rows & 0xffffu),                         // tile_dim1 | tile_dim2=0
                 (int)strideu,                                  // tensor_dim0_stride[31:0]
                 0, 0 };
    i32x4 gz = { 0, 0, 0, 0 };
#if defined(__clang_major__) && (__clang_major__ >= 23)
    i32x8 gz8 = { 0, 0, 0, 0, 0, 0, 0, 0 };
    __builtin_amdgcn_tensor_load_to_lds(g0, g1, gz, gz, gz8, 0);
#else
    __builtin_amdgcn_tensor_load_to_lds(g0, g1, gz, gz, 0);
#endif
}

// ---------------------------------------------------------------------------
// x (B,C,N) f32 -> xT (B,N,C) bf16, LDS-tiled transpose (coalesced both sides)
// ---------------------------------------------------------------------------
__global__ __launch_bounds__(256) void transpose_to_bf16(
    const float* __restrict__ x, unsigned short* __restrict__ xT, int C, int Npix)
{
    __shared__ float tile[32][33];
    const int b  = blockIdx.z;
    const int p0 = blockIdx.x * 32;
    const int c0 = blockIdx.y * 32;
    const int tx = threadIdx.x & 31;
    const int ty = threadIdx.x >> 5;         // 0..7
    const float* xb = x + (size_t)b * C * Npix;
    #pragma unroll
    for (int i = 0; i < 4; ++i)
        tile[ty + 8*i][tx] = xb[(size_t)(c0 + ty + 8*i) * Npix + p0 + tx];
    __syncthreads();
    unsigned short* ob = xT + (size_t)b * Npix * C;
    #pragma unroll
    for (int i = 0; i < 4; ++i)
        ob[(size_t)(p0 + ty + 8*i) * C + c0 + tx] = f2bf(tile[tx][ty + 8*i]);
}

__global__ void f32_to_bf16(const float* __restrict__ in,
                            unsigned short* __restrict__ out, int n)
{
    int i = blockIdx.x * blockDim.x + threadIdx.x;
    if (i < n) out[i] = f2bf(in[i]);
}

// BN folding: s = g/sqrt(v+eps), t = b - m*s
__global__ void bn_prep(const float* __restrict__ g, const float* __restrict__ bb,
                        const float* __restrict__ mu, const float* __restrict__ var,
                        float* __restrict__ s, float* __restrict__ t, int c)
{
    int i = blockIdx.x * blockDim.x + threadIdx.x;
    if (i < c) { float sv = g[i] * rsqrtf(var[i] + 1e-5f); s[i] = sv; t[i] = bb[i] - mu[i] * sv; }
}

// ---------------------------------------------------------------------------
// conv1x1 GEMM: Out = act(W(CoxCi) @ Xp^T), Xp pixel-major (B,N,Ci) bf16.
// Block = 128 threads (4 waves); wave -> 16(M) x 64(pixels) output strip.
// MODE 0: pixel-major bf16 (B,N,Co)     (feeds next GEMM / attention q,k)
// MODE 1: channel-major bf16 (B,Co,N)   (v for attention)
// MODE 2: channel-major f32  (B,Co,N)   (final output)
// ---------------------------------------------------------------------------
template<int CI, int CO, int MODE, bool BNRELU>
__global__ __launch_bounds__(128) void gemm1x1(
    const unsigned short* __restrict__ Xp,
    const unsigned short* __restrict__ Wb,
    const float* __restrict__ bns, const float* __restrict__ bnt,
    void* __restrict__ OutP, float outScale)
{
    const int lane = threadIdx.x & 31;
    const int wave = threadIdx.x >> 5;
    const int half = lane >> 4;
    const int ln   = lane & 15;
    const int m0   = blockIdx.y * 64 + wave * 16;
    const int p0   = blockIdx.x * 64;
    const int b    = blockIdx.z;
    const unsigned short* xb = Xp + (size_t)b * NPIX * CI;

    f32x8 acc[4] = {};

    #pragma unroll 4
    for (int s = 0; s < CI / 32; ++s) {
        FragU a;   // A = weight tile 16x32: lane row = ln, regs0-3 k=8h.., regs4-7 k=16+8h..
        const unsigned short* ap = Wb + (size_t)(m0 + ln) * CI + 32*s + 8*half;
        a.q[0] = *(const uint4*)ap;
        a.q[1] = *(const uint4*)(ap + 16);
        #pragma unroll
        for (int ct = 0; ct < 4; ++ct) {
            FragU bb;  // B = activation tile 32x16: lane col = ln, regs0-3 k=8h.., regs4-7 k=16+8h..
            const unsigned short* bp = xb + (size_t)(p0 + ct*16 + ln) * CI + 32*s + 8*half;
            bb.q[0] = *(const uint4*)bp;
            bb.q[1] = *(const uint4*)(bp + 16);
            acc[ct] = wmma_bf16(a.m, bb.m, acc[ct]);
        }
    }

    #pragma unroll
    for (int ct = 0; ct < 4; ++ct) {
        const int pix = p0 + ct*16 + ln;
        #pragma unroll
        for (int r = 0; r < 8; r += 2) {
            const int ch0 = m0 + 8*half + r;          // D: row = r + 8*half
            float v0 = acc[ct][r], v1 = acc[ct][r+1];
            if constexpr (BNRELU) {
                v0 = fmaxf(v0 * bns[ch0]     + bnt[ch0],     0.0f);
                v1 = fmaxf(v1 * bns[ch0 + 1] + bnt[ch0 + 1], 0.0f);
            }
            v0 *= outScale; v1 *= outScale;
            if constexpr (MODE == 0) {
                unsigned short* ob = (unsigned short*)OutP + (size_t)b * NPIX * CO;
                *(unsigned int*)(ob + (size_t)pix * CO + ch0) = packbf(v0, v1);
            } else if constexpr (MODE == 1) {
                unsigned short* ob = (unsigned short*)OutP + (size_t)b * CO * NPIX;
                ob[(size_t)ch0       * NPIX + pix] = f2bf(v0);
                ob[(size_t)(ch0 + 1) * NPIX + pix] = f2bf(v1);
            } else {
                float* ob = (float*)OutP + (size_t)b * CO * NPIX;
                ob[(size_t)ch0       * NPIX + pix] = v0;
                ob[(size_t)(ch0 + 1) * NPIX + pix] = v1;
            }
        }
    }
}

// ---------------------------------------------------------------------------
// Flash attention, TDM-staged: block = 4 waves = 64 queries; K/V 32-key tiles
// are DMA'd into a double-buffered LDS stage by the Tensor Data Mover (issued
// by wave 0, tracked with TENSORcnt) while all 4 waves run WMMA on the
// previous tile.  S^T = K^T Q; online softmax per query column (half-wave
// combine via ds_swizzle SWAPX16); ctx^T += V * P^T.
// ---------------------------------------------------------------------------
__global__ __launch_bounds__(128) void flash_attn(
    const unsigned short* __restrict__ qT,   // (B,N,CK) bf16, pre-scaled by 1/16
    const unsigned short* __restrict__ kT,   // (B,N,CK) bf16
    const unsigned short* __restrict__ V,    // (B,CV,N) bf16 channel-major
    unsigned short* __restrict__ ctxT)       // (B,N,CV) bf16
{
    __shared__ __align__(16) unsigned short ksm[2][KTILE * CK];  // 2 x 16KB
    __shared__ __align__(16) unsigned short vsm[2][CV * KTILE];  // 2 x 16KB

    const int tid  = threadIdx.x;
    const int lane = tid & 31;
    const int wave = tid >> 5;
    const int half = lane >> 4;
    const int ln   = lane & 15;
    const int b    = blockIdx.y;
    const int n0   = blockIdx.x * 64 + wave * 16;

    const unsigned short* qb  = qT + (size_t)b * NPIX * CK;
    const unsigned short* kbp = kT + (size_t)b * NPIX * CK;
    const unsigned short* vb  = V  + (size_t)b * CV * NPIX;

    // Q as B operand of S^T: col = query, k = channel (regs0-3 k=8h.., regs4-7 k=16+8h..); per wave
    FragU qf[8];
    #pragma unroll
    for (int s = 0; s < 8; ++s) {
        const unsigned short* qp = qb + (size_t)(n0 + ln) * CK + 32*s + 8*half;
        qf[s].q[0] = *(const uint4*)qp;
        qf[s].q[1] = *(const uint4*)(qp + 16);
    }

    f32x8 O[16] = {};                 // ctx^T tiles: 16 x (16ch x 16queries)
    float mrow = -3.0e38f, lrow = 0.0f;

    const unsigned kbase = (unsigned)(uintptr_t)&ksm[0][0];   // LDS byte offsets
    const unsigned vbase = (unsigned)(uintptr_t)&vsm[0][0];

    // prologue: DMA tile 0 into buffer 0
    if (wave == 0) {
        tdm_load_2d(kbase, kbp, CK * 2 / 8, KTILE, CK * 2 / 8);       // 32 rows x 512B
        tdm_load_2d(vbase, vb,  KTILE * 2 / 8, CV, NPIX * 2 / 8);     // 256 rows x 64B
    }

    for (int kb = 0; kb < NPIX; kb += KTILE) {
        const int cur = (kb / KTILE) & 1;
        if (wave == 0) __builtin_amdgcn_s_wait_tensorcnt(0);
        __syncthreads();              // staged tile visible to all waves

        // prefetch next tile into the other buffer while we compute
        if (wave == 0 && kb + KTILE < NPIX) {
            const int nxt = cur ^ 1;
            tdm_load_2d(kbase + (unsigned)nxt * KTILE * CK * 2,
                        kbp + (size_t)(kb + KTILE) * CK, CK * 2 / 8, KTILE, CK * 2 / 8);
            tdm_load_2d(vbase + (unsigned)nxt * CV * KTILE * 2,
                        vb + (kb + KTILE), KTILE * 2 / 8, CV, NPIX * 2 / 8);
        }

        const unsigned short* ks = ksm[cur];
        const unsigned short* vs = vsm[cur];

        f32x8 t0 = {}, t1 = {};       // S^T tiles: keys kb..+15 / kb+16..+31
        #pragma unroll
        for (int s = 0; s < 8; ++s) {
            FragU a0, a1;             // A = K^T tile from LDS: row = key, k = channel
            const unsigned short* a0p = ks + (ln)      * CK + 32*s + 8*half;
            const unsigned short* a1p = ks + (16 + ln) * CK + 32*s + 8*half;
            a0.q[0] = *(const uint4*)a0p; a0.q[1] = *(const uint4*)(a0p + 16);
            a1.q[0] = *(const uint4*)a1p; a1.q[1] = *(const uint4*)(a1p + 16);
            t0 = wmma_bf16(a0.m, qf[s].m, t0);
            t1 = wmma_bf16(a1.m, qf[s].m, t1);
        }

        // ---- online softmax for this lane's query column (32 key scores) ----
        float mx = -3.0e38f;
        #pragma unroll
        for (int r = 0; r < 8; ++r) { mx = fmaxf(mx, t0[r]); mx = fmaxf(mx, t1[r]); }
        mx = fmaxf(mx, swapx16(mx));                      // combine half-pairs
        const float mnew  = fmaxf(mrow, mx);
        const float alpha = __expf(mrow - mnew);
        float p0v[8], p1v[8], sum = 0.0f;
        #pragma unroll
        for (int r = 0; r < 8; ++r) {
            p0v[r] = __expf(t0[r] - mnew);
            p1v[r] = __expf(t1[r] - mnew);
            sum += p0v[r] + p1v[r];
        }
        sum += swapx16(sum);
        lrow = lrow * alpha + sum;
        mrow = mnew;
        #pragma unroll
        for (int t = 0; t < 16; ++t) O[t] *= alpha;

        // ---- re-lane exp(S^T) (C layout) into B layout P^T (32x16 bf16) ----
        float sp0[8], sp1[8];
        #pragma unroll
        for (int r = 0; r < 8; ++r) { sp0[r] = swapx16(p0v[r]); sp1[r] = swapx16(p1v[r]); }
        PFragU pf;
        #pragma unroll
        for (int j = 0; j < 4; ++j) {
            // regs 0..3: keys 2j,2j+1 (+16 for upper half-lanes)
            float x0 = half ? sp1[2*j]     : p0v[2*j];
            float x1 = half ? sp1[2*j + 1] : p0v[2*j + 1];
            pf.u[j] = packbf(x0, x1);
            // regs 4..7: keys 8+2j,9+2j (+16 for upper half-lanes)
            float y0 = half ? p1v[2*j]     : sp0[2*j];
            float y1 = half ? p1v[2*j + 1] : sp0[2*j + 1];
            pf.u[j + 4] = packbf(y0, y1);
        }

        // ---- ctx^T += V(16ch x 32keys) * P^T(32keys x 16queries) ----
        #pragma unroll
        for (int t = 0; t < 16; ++t) {
            FragU vf;   // A operand from LDS: row = channel, k = key
            const unsigned short* vp = vs + (size_t)(16*t + ln) * KTILE + 16*half;
            vf.q[0] = *(const uint4*)vp; vf.q[1] = *(const uint4*)(vp + 8);
            O[t] = wmma_bf16(vf.m, pf.m, O[t]);
        }
    }

    const float inv = 1.0f / lrow;
    unsigned short* ob = ctxT + (size_t)b * NPIX * CV;
    const int pix = n0 + ln;
    #pragma unroll
    for (int t = 0; t < 16; ++t) {
        #pragma unroll
        for (int r = 0; r < 8; r += 2) {
            const int ch = 16*t + 8*half + r;
            *(unsigned int*)(ob + (size_t)pix * CV + ch) =
                packbf(O[t][r] * inv, O[t][r + 1] * inv);
        }
    }
}

// ---------------------------------------------------------------------------
// Host launch
// ---------------------------------------------------------------------------
extern "C" void kernel_launch(void* const* d_in, const int* in_sizes, int n_in,
                              void* d_out, int out_size, void* d_ws, size_t ws_size,
                              hipStream_t stream)
{
    (void)in_sizes; (void)n_in; (void)out_size; (void)ws_size;
    const float* x   = (const float*)d_in[0];
    const float* kW1 = (const float*)d_in[1];
    const float* kG1 = (const float*)d_in[2];  const float* kB1 = (const float*)d_in[3];
    const float* kM1 = (const float*)d_in[4];  const float* kV1 = (const float*)d_in[5];
    const float* kW2 = (const float*)d_in[6];
    const float* kG2 = (const float*)d_in[7];  const float* kB2 = (const float*)d_in[8];
    const float* kM2 = (const float*)d_in[9];  const float* kV2 = (const float*)d_in[10];
    const float* qW1 = (const float*)d_in[11];
    const float* qG1 = (const float*)d_in[12]; const float* qB1 = (const float*)d_in[13];
    const float* qM1 = (const float*)d_in[14]; const float* qV1 = (const float*)d_in[15];
    const float* qW2 = (const float*)d_in[16];
    const float* qG2 = (const float*)d_in[17]; const float* qB2 = (const float*)d_in[18];
    const float* qM2 = (const float*)d_in[19]; const float* qV2 = (const float*)d_in[20];
    const float* vW  = (const float*)d_in[21];
    const float* wW  = (const float*)d_in[22];
    const float* wG  = (const float*)d_in[23]; const float* wB  = (const float*)d_in[24];
    const float* wM  = (const float*)d_in[25]; const float* wV  = (const float*)d_in[26];
    float* out = (float*)d_out;

    char* ws = (char*)d_ws;
    size_t off = 0;
    auto alloc = [&](size_t bytes) -> void* {
        void* p = (void*)(ws + off);
        off += (bytes + 255) & ~(size_t)255;
        return p;
    };
    unsigned short* xT   = (unsigned short*)alloc((size_t)B_ * NPIX * CIN * 2);
    unsigned short* wq1b = (unsigned short*)alloc((size_t)CK * CIN * 2);
    unsigned short* wq2b = (unsigned short*)alloc((size_t)CK * CK  * 2);
    unsigned short* wk1b = (unsigned short*)alloc((size_t)CK * CIN * 2);
    unsigned short* wk2b = (unsigned short*)alloc((size_t)CK * CK  * 2);
    unsigned short* wvb  = (unsigned short*)alloc((size_t)CV * CIN * 2);
    unsigned short* wwb  = (unsigned short*)alloc((size_t)COUT * CV * 2);
    float* sq1 = (float*)alloc(CK * 4);   float* tq1 = (float*)alloc(CK * 4);
    float* sq2 = (float*)alloc(CK * 4);   float* tq2 = (float*)alloc(CK * 4);
    float* sk1 = (float*)alloc(CK * 4);   float* tk1 = (float*)alloc(CK * 4);
    float* sk2 = (float*)alloc(CK * 4);   float* tk2 = (float*)alloc(CK * 4);
    float* sw  = (float*)alloc(COUT * 4); float* tw  = (float*)alloc(COUT * 4);
    unsigned short* t1b  = (unsigned short*)alloc((size_t)B_ * NPIX * CK * 2);
    unsigned short* qTb  = (unsigned short*)alloc((size_t)B_ * NPIX * CK * 2);
    unsigned short* kTb  = (unsigned short*)alloc((size_t)B_ * NPIX * CK * 2);
    unsigned short* vbuf = (unsigned short*)alloc((size_t)B_ * CV * NPIX * 2);
    unsigned short* ctxb = (unsigned short*)alloc((size_t)B_ * NPIX * CV * 2);

    // 0) x -> pixel-major bf16
    transpose_to_bf16<<<dim3(NPIX/32, CIN/32, B_), 256, 0, stream>>>(x, xT, CIN, NPIX);

    // 1) weights -> bf16, BN folding
    auto cvt = [&](const float* w, unsigned short* o, int n) {
        f32_to_bf16<<<(n + 255) / 256, 256, 0, stream>>>(w, o, n);
    };
    cvt(qW1, wq1b, CK * CIN); cvt(qW2, wq2b, CK * CK);
    cvt(kW1, wk1b, CK * CIN); cvt(kW2, wk2b, CK * CK);
    cvt(vW,  wvb,  CV * CIN); cvt(wW,  wwb,  COUT * CV);
    bn_prep<<<1, 256, 0, stream>>>(qG1, qB1, qM1, qV1, sq1, tq1, CK);
    bn_prep<<<1, 256, 0, stream>>>(qG2, qB2, qM2, qV2, sq2, tq2, CK);
    bn_prep<<<1, 256, 0, stream>>>(kG1, kB1, kM1, kV1, sk1, tk1, CK);
    bn_prep<<<1, 256, 0, stream>>>(kG2, kB2, kM2, kV2, sk2, tk2, CK);
    bn_prep<<<2, 256, 0, stream>>>(wG,  wB,  wM,  wV,  sw,  tw,  COUT);

    // 2) q path (softmax scale Ck^-0.5 = 1/16 folded into q store)
    gemm1x1<CIN, CK, 0, true><<<dim3(NPIX/64, CK/64, B_), 128, 0, stream>>>(
        xT, wq1b, sq1, tq1, t1b, 1.0f);
    gemm1x1<CK, CK, 0, true><<<dim3(NPIX/64, CK/64, B_), 128, 0, stream>>>(
        t1b, wq2b, sq2, tq2, qTb, 0.0625f);
    // 3) k path
    gemm1x1<CIN, CK, 0, true><<<dim3(NPIX/64, CK/64, B_), 128, 0, stream>>>(
        xT, wk1b, sk1, tk1, t1b, 1.0f);
    gemm1x1<CK, CK, 0, true><<<dim3(NPIX/64, CK/64, B_), 128, 0, stream>>>(
        t1b, wk2b, sk2, tk2, kTb, 1.0f);
    // 4) v (channel-major, no BN)
    gemm1x1<CIN, CV, 1, false><<<dim3(NPIX/64, CV/64, B_), 128, 0, stream>>>(
        xT, wvb, nullptr, nullptr, vbuf, 1.0f);
    // 5) flash attention (TDM-staged K/V) -> ctx^T (pixel-major bf16)
    flash_attn<<<dim3(NPIX/64, B_), 128, 0, stream>>>(qTb, kTb, vbuf, ctxb);
    // 6) output projection + BN + ReLU -> f32 (B,Cout,H,W)
    gemm1x1<CV, COUT, 2, true><<<dim3(NPIX/64, COUT/64, B_), 128, 0, stream>>>(
        ctxb, wwb, sw, tw, out, 1.0f);
}
